// VanillaRNN_44470091383029
// MI455X (gfx1250) — matrix operand order repaired and checked by reading, not compiled, not run on hardware
//
#include <hip/hip_runtime.h>
#include <hip/hip_bf16.h>

// ---------------------------------------------------------------------------
// 2-layer vanilla RNN (B=64, T=1024, E=128, H=512, V=256), CDNA5 gfx1250.
//   1) embed gather -> bf16 [T][B][E]
//   2) weights -> bf16, pre-swizzled into WMMA B-fragment layout
//   3) persistent single-workgroup (32 waves) recurrence kernel, h0/h1 in LDS,
//      e_t double-buffered in LDS via TDM tensor_load_to_lds (async, hidden
//      behind WMMA compute), v_wmma_f32_16x16x32_bf16 for all matmuls.
//      A-fragments for all 4 M-tiles are loaded before the WMMA burst so the
//      ds loads clause up and WMMAs issue back-to-back.
//   4) whole-GPU parallel FC GEMM kernel (WMMA) producing fp32 logits
// ---------------------------------------------------------------------------

typedef __bf16 bf16_t;
typedef __attribute__((ext_vector_type(16))) __bf16 v16bf;
typedef __attribute__((ext_vector_type(8)))  __bf16 v8bf;
typedef __attribute__((ext_vector_type(8)))  float  v8f;
typedef unsigned int u32;
typedef unsigned long long u64;
typedef __attribute__((ext_vector_type(4))) u32 u32x4;
typedef __attribute__((ext_vector_type(8))) u32 u32x8;

#define RNN_B 64
#define RNN_T 1024
#define RNN_E 128
#define RNN_H 512
#define RNN_V 256

// ---------------------------------------------------------------------------
// WMMA helpers
// ---------------------------------------------------------------------------
__device__ __forceinline__ v8f wmma_bf16(v16bf a, v16bf b, v8f c) {
  return __builtin_amdgcn_wmma_f32_16x16x32_bf16(
      /*neg_a=*/false, a, /*neg_b=*/false, b,
      /*c_mod=*/(short)0, c, /*reuse_a=*/false, /*reuse_b=*/false);
}

// A fragment: 16x32 tile of a row-major MxK bf16 matrix (row stride ld elems).
__device__ __forceinline__ v16bf load_a_frag(const bf16_t* base, int ld,
                                             int mrow, int k0, int lane) {
  int row = mrow + (lane & 15);
  int kb  = k0 + ((lane >> 4) << 4);
  const v8bf* p0 = (const v8bf*)(base + row * ld + kb);
  const v8bf* p1 = (const v8bf*)(base + row * ld + kb + 8);
  v8bf lo = *p0;
  v8bf hi = *p1;
  v16bf a;
#pragma unroll
  for (int i = 0; i < 8; ++i) { a[i] = lo[i]; a[i + 8] = hi[i]; }
  return a;
}

// B fragment from pre-swizzled weights: tile (kt,nt) = 512 contiguous bf16,
// lane-major (32 lanes x 16 elems): one 32-byte load per lane.
__device__ __forceinline__ v16bf load_b_frag(const bf16_t* swz, int numTilesN,
                                             int kt, int nt, int lane) {
  const v16bf* p =
      (const v16bf*)(swz + (((kt * numTilesN + nt) << 9) + (lane << 4)));
  return *p;
}

// One K-chunk for a 64-row A panel: load all 4 M-tile A-fragments, then run
// the 4 WMMAs back-to-back (single ds wait, XDL co-execution friendly).
__device__ __forceinline__ void gemm_chunk(const bf16_t* apanel, int ld,
                                           int k0, v16bf bfrag, v8f acc[4],
                                           int lane) {
  v16bf a0 = load_a_frag(apanel, ld, 0,  k0, lane);
  v16bf a1 = load_a_frag(apanel, ld, 16, k0, lane);
  v16bf a2 = load_a_frag(apanel, ld, 32, k0, lane);
  v16bf a3 = load_a_frag(apanel, ld, 48, k0, lane);
  acc[0] = wmma_bf16(a0, bfrag, acc[0]);
  acc[1] = wmma_bf16(a1, bfrag, acc[1]);
  acc[2] = wmma_bf16(a2, bfrag, acc[2]);
  acc[3] = wmma_bf16(a3, bfrag, acc[3]);
}

// ---------------------------------------------------------------------------
// TDM: async DMA of a contiguous run of `ndwords` dwords from global -> LDS.
// Builds a D# (group0 + group1) in SGPRs and issues tensor_load_to_lds via
// inline asm (portable across ROCm-7.2 / therock builtin-arity mismatch).
// Completion is tracked with TENSORcnt.
// ---------------------------------------------------------------------------
__device__ __forceinline__ void tdm_load_1d(u32 lds_byte, u64 gaddr,
                                            u32 ndwords) {
  u32x4 g0;
  g0[0] = 1u;                                              // count=1
  g0[1] = lds_byte;                                        // lds_addr
  g0[2] = (u32)gaddr;                                      // global_addr lo
  g0[3] = ((u32)(gaddr >> 32) & 0x01FFFFFFu) | (2u << 30); // hi | type=2
  u32x8 g1;
  g1[0] = (2u << 16);                     // mask=0, data_size=4B, no flags
  g1[1] = (ndwords & 0xFFFFu) << 16;      // tensor_dim0[15:0]
  g1[2] = (ndwords >> 16) | (1u << 16);   // tensor_dim0[31:16] | tensor_dim1=1
  g1[3] = (ndwords & 0xFFFFu) << 16;      // tile_dim0 = ndwords
  g1[4] = 1u;                             // tile_dim1=1, tile_dim2=0
  g1[5] = ndwords;                        // tensor_dim0_stride[31:0]
  g1[6] = (ndwords & 0xFFFFu) << 16;      // stride hi=0 | dim1_stride[15:0]
  g1[7] = (ndwords >> 16);                // dim1_stride[47:16]
  asm volatile("tensor_load_to_lds %0, %1" :: "s"(g0), "s"(g1) : "memory");
}

// ---------------------------------------------------------------------------
// Kernel 1: embedding gather + fp32->bf16, layout [T][B][E]
// ---------------------------------------------------------------------------
__global__ void rnn_embed_kernel(const long long* __restrict__ x,
                                 const float* __restrict__ emb,
                                 bf16_t* __restrict__ e) {
  int idx = blockIdx.x * 256 + threadIdx.x;   // < T*B*E
  int t = idx / (RNN_B * RNN_E);
  int r = idx % (RNN_B * RNN_E);
  int b = r >> 7;
  int k = r & (RNN_E - 1);
  long long tok = x[(long long)b * RNN_T + t];
  e[idx] = (bf16_t)emb[tok * RNN_E + k];
}

// ---------------------------------------------------------------------------
// Kernel 2: weight convert. src = W [N][K] fp32; logical B = W^T (K x N),
// stored as swizzled WMMA B-fragments (bf16).
// ---------------------------------------------------------------------------
__global__ void rnn_convert_wswz(const float* __restrict__ src,
                                 bf16_t* __restrict__ dst, int N, int K) {
  int idx = blockIdx.x * 256 + threadIdx.x;
  if (idx >= N * K) return;
  int n = idx / K;
  int k = idx - n * K;
  int kt = k >> 5;
  int nt = n >> 4;
  int lane = (n & 15) + (((k >> 4) & 1) << 4);
  int el = k & 15;
  dst[(((kt * (N >> 4) + nt) << 9) + (lane << 4)) + el] = (bf16_t)src[idx];
}

// ---------------------------------------------------------------------------
// Kernel 3: persistent recurrence. 32 waves; wave w owns hidden N-tile w.
// h0/h1 in LDS bf16 [64][512]; e_t double-buffered in LDS, filled by TDM.
// ---------------------------------------------------------------------------
__global__ __launch_bounds__(1024) void rnn_recurrence_kernel(
    const bf16_t* __restrict__ e,       // [T][B][E] bf16
    const bf16_t* __restrict__ wih0s,   // swz K=E,  N=H
    const bf16_t* __restrict__ whh0s,   // swz K=H,  N=H
    const bf16_t* __restrict__ wih1s,   // swz K=H,  N=H
    const bf16_t* __restrict__ whh1s,   // swz K=H,  N=H
    const float* __restrict__ bih0, const float* __restrict__ bhh0,
    const float* __restrict__ bih1, const float* __restrict__ bhh1,
    bf16_t* __restrict__ h1buf)         // [T][B][H] bf16
{
  __shared__ bf16_t h0[RNN_B * RNN_H];
  __shared__ bf16_t h1[RNN_B * RNN_H];
  __shared__ bf16_t ebuf[2][RNN_B * RNN_E];   // double-buffered e_t (16KB ea.)

  const int tid  = threadIdx.x;
  const int lane = tid & 31;
  const int wave = tid >> 5;
  const int hi   = lane >> 4;
  const int nt   = wave;
  const int ncol = (nt << 4) + (lane & 15);

  constexpr u32 ETILE_DW = (RNN_B * RNN_E * 2) / 4;   // 4096 dwords / step

  for (int i = tid; i < RNN_B * RNN_H; i += 1024) {
    h0[i] = (bf16_t)0.0f;
    h1[i] = (bf16_t)0.0f;
  }

  // prologue: DMA e_0 into buffer 0 (wave 0 issues; TENSORcnt tracks it)
  if (wave == 0) {
    tdm_load_1d((u32)(size_t)(&ebuf[0][0]), (u64)(size_t)e, ETILE_DW);
    __builtin_amdgcn_s_wait_tensorcnt(0);
  }
  __syncthreads();

  const float bias0 = bih0[ncol] + bhh0[ncol];
  const float bias1 = bih1[ncol] + bhh1[ncol];

  for (int t = 0; t < RNN_T; ++t) {
    const bf16_t* et = ebuf[t & 1];

    // async prefetch of e_{t+1} into the other buffer, hidden behind compute
    if (wave == 0 && (t + 1) < RNN_T) {
      tdm_load_1d((u32)(size_t)(&ebuf[(t + 1) & 1][0]),
                  (u64)(size_t)(e + (size_t)(t + 1) * (RNN_B * RNN_E)),
                  ETILE_DW);
    }

    // ---- layer 0: acc = e_t @ w_ih0^T + h0 @ w_hh0^T + biases -------------
    v8f acc[4];
#pragma unroll
    for (int m = 0; m < 4; ++m)
#pragma unroll
      for (int i = 0; i < 8; ++i) acc[m][i] = bias0;

    for (int kc = 0; kc < RNN_E / 32; ++kc) {
      v16bf bfrag = load_b_frag(wih0s, RNN_H / 16, kc, nt, lane);
      gemm_chunk(et, RNN_E, kc << 5, bfrag, acc, lane);
    }
    for (int kc = 0; kc < RNN_H / 32; ++kc) {
      v16bf bfrag = load_b_frag(whh0s, RNN_H / 16, kc, nt, lane);
      gemm_chunk(h0, RNN_H, kc << 5, bfrag, acc, lane);
    }
    __syncthreads();   // everyone done reading old h0

#pragma unroll
    for (int m = 0; m < 4; ++m)
#pragma unroll
      for (int i = 0; i < 8; ++i) {
        int row = (m << 4) + i + (hi << 3);
        h0[row * RNN_H + ncol] = (bf16_t)tanhf(acc[m][i]);
      }
    __syncthreads();   // new h0 visible

    // ---- layer 1: acc = h0 @ w_ih1^T + h1 @ w_hh1^T + biases --------------
#pragma unroll
    for (int m = 0; m < 4; ++m)
#pragma unroll
      for (int i = 0; i < 8; ++i) acc[m][i] = bias1;

    for (int kc = 0; kc < RNN_H / 32; ++kc) {
      v16bf b0 = load_b_frag(wih1s, RNN_H / 16, kc, nt, lane);
      gemm_chunk(h0, RNN_H, kc << 5, b0, acc, lane);
      v16bf b1 = load_b_frag(whh1s, RNN_H / 16, kc, nt, lane);
      gemm_chunk(h1, RNN_H, kc << 5, b1, acc, lane);
    }
    __syncthreads();   // everyone done reading old h1

    bf16_t* ht = h1buf + (size_t)t * (RNN_B * RNN_H);
#pragma unroll
    for (int m = 0; m < 4; ++m)
#pragma unroll
      for (int i = 0; i < 8; ++i) {
        int row = (m << 4) + i + (hi << 3);
        bf16_t v = (bf16_t)tanhf(acc[m][i]);
        h1[row * RNN_H + ncol] = v;
        ht[row * RNN_H + ncol] = v;       // stream out for FC kernel
      }

    // TDM prefetch of e_{t+1} must land before anyone reads it next step
    if (wave == 0) __builtin_amdgcn_s_wait_tensorcnt(0);
    __syncthreads();   // new h1 + e_{t+1} visible
  }
}

// ---------------------------------------------------------------------------
// Kernel 4: parallel FC GEMM. h1buf [T*B][H] bf16 (row r = t*64 + b),
// logits[b][t][v] = h1 @ fc_w^T + fc_b, fp32 out. One 16x16 tile per wave.
// ---------------------------------------------------------------------------
__global__ __launch_bounds__(256) void rnn_fc_kernel(
    const bf16_t* __restrict__ h1buf,
    const bf16_t* __restrict__ fcs,     // swz K=H, N=V
    const float* __restrict__ fcb,
    float* __restrict__ out) {
  const int gw    = (blockIdx.x * 256 + threadIdx.x) >> 5;
  const int lane  = threadIdx.x & 31;
  const int mt    = gw >> 4;
  const int ntile = gw & 15;
  const int mrow  = mt << 4;
  const int ncol  = (ntile << 4) + (lane & 15);
  const int hi    = lane >> 4;

  float bias = fcb[ncol];
  v8f acc;
#pragma unroll
  for (int i = 0; i < 8; ++i) acc[i] = bias;

  for (int kc = 0; kc < RNN_H / 32; ++kc) {
    v16bf bfrag = load_b_frag(fcs, RNN_V / 16, kc, ntile, lane);
    v16bf af    = load_a_frag(h1buf, RNN_H, mrow, kc << 5, lane);
    acc = wmma_bf16(af, bfrag, acc);
  }

#pragma unroll
  for (int i = 0; i < 8; ++i) {
    int r  = mrow + i + (hi << 3);
    int tt = r >> 6;
    int b  = r & 63;
    out[((size_t)b * RNN_T + tt) * RNN_V + ncol] = acc[i];
  }
}

// ---------------------------------------------------------------------------
// Launch
// ---------------------------------------------------------------------------
extern "C" void kernel_launch(void* const* d_in, const int* in_sizes, int n_in,
                              void* d_out, int out_size, void* d_ws,
                              size_t ws_size, hipStream_t stream) {
  const long long* x    = (const long long*)d_in[0];
  const float* emb      = (const float*)d_in[1];
  const float* w_ih0    = (const float*)d_in[2];
  const float* b_ih0    = (const float*)d_in[3];
  const float* w_hh0    = (const float*)d_in[4];
  const float* b_hh0    = (const float*)d_in[5];
  const float* w_ih1    = (const float*)d_in[6];
  const float* b_ih1    = (const float*)d_in[7];
  const float* w_hh1    = (const float*)d_in[8];
  const float* b_hh1    = (const float*)d_in[9];
  const float* fc_w     = (const float*)d_in[10];
  const float* fc_b     = (const float*)d_in[11];
  float* out            = (float*)d_out;

  constexpr size_t OFF_E     = 0;
  constexpr size_t OFF_H1    = OFF_E  + (size_t)RNN_T * RNN_B * RNN_E * 2;
  constexpr size_t OFF_WIH0  = OFF_H1 + (size_t)RNN_T * RNN_B * RNN_H * 2;
  constexpr size_t OFF_WHH0  = OFF_WIH0 + (size_t)RNN_H * RNN_E * 2;
  constexpr size_t OFF_WIH1  = OFF_WHH0 + (size_t)RNN_H * RNN_H * 2;
  constexpr size_t OFF_WHH1  = OFF_WIH1 + (size_t)RNN_H * RNN_H * 2;
  constexpr size_t OFF_FCW   = OFF_WHH1 + (size_t)RNN_H * RNN_H * 2;

  char* ws = (char*)d_ws;
  bf16_t* e_bf   = (bf16_t*)(ws + OFF_E);
  bf16_t* h1buf  = (bf16_t*)(ws + OFF_H1);
  bf16_t* wih0s  = (bf16_t*)(ws + OFF_WIH0);
  bf16_t* whh0s  = (bf16_t*)(ws + OFF_WHH0);
  bf16_t* wih1s  = (bf16_t*)(ws + OFF_WIH1);
  bf16_t* whh1s  = (bf16_t*)(ws + OFF_WHH1);
  bf16_t* fcs    = (bf16_t*)(ws + OFF_FCW);

  rnn_convert_wswz<<<(RNN_H * RNN_E + 255) / 256, 256, 0, stream>>>(
      w_ih0, wih0s, RNN_H, RNN_E);
  rnn_convert_wswz<<<(RNN_H * RNN_H + 255) / 256, 256, 0, stream>>>(
      w_hh0, whh0s, RNN_H, RNN_H);
  rnn_convert_wswz<<<(RNN_H * RNN_H + 255) / 256, 256, 0, stream>>>(
      w_ih1, wih1s, RNN_H, RNN_H);
  rnn_convert_wswz<<<(RNN_H * RNN_H + 255) / 256, 256, 0, stream>>>(
      w_hh1, whh1s, RNN_H, RNN_H);
  rnn_convert_wswz<<<(RNN_V * RNN_H + 255) / 256, 256, 0, stream>>>(
      fc_w, fcs, RNN_V, RNN_H);

  rnn_embed_kernel<<<(RNN_T * RNN_B * RNN_E) / 256, 256, 0, stream>>>(
      x, emb, e_bf);

  rnn_recurrence_kernel<<<1, 1024, 0, stream>>>(
      e_bf, wih0s, whh0s, wih1s, whh1s,
      b_ih0, b_hh0, b_ih1, b_hh1, h1buf);

  rnn_fc_kernel<<<(65536 * 32) / 256, 256, 0, stream>>>(h1buf, fcs, fc_b, out);
}
